// VoiceChanger_21981642621205
// MI455X (gfx1250) — matrix operand order repaired and checked by reading, not compiled, not run on hardware
//
#include <hip/hip_runtime.h>
#include <math.h>

typedef __attribute__((ext_vector_type(16))) _Float16 v16h;
typedef __attribute__((ext_vector_type(4)))  _Float16 v4h;
typedef __attribute__((ext_vector_type(8)))  float    v8f;

#define N1   2048
#define N2   65536
#define DIM  1024
#define TOPK 4
#define NROW 128                  // matching rows staged per block
#define NBLK (N2 / NROW)          // 512 blocks
#define LDS_STRIDE (DIM + 16)     // padded LDS row stride (halfs)
#define CAND (NBLK * TOPK)        // 2048 candidates per query

// ---------------------------------------------------------------------------
// Kernel 1: inverse L2 norm of each matching row (1/|m|). 8 rows per block,
// one wave (32 lanes) per row.
// ---------------------------------------------------------------------------
__global__ void __launch_bounds__(256)
norms_kernel(const float* __restrict__ M, float* __restrict__ invn) {
  const int wave = threadIdx.x >> 5;
  const int lane = threadIdx.x & 31;
  const int row  = blockIdx.x * 8 + wave;
  const float4* p = (const float4*)(M + (size_t)row * DIM);
  float s = 0.f;
#pragma unroll
  for (int i = 0; i < 8; ++i) {
    float4 v = p[lane + i * 32];
    s += v.x * v.x + v.y * v.y + v.z * v.z + v.w * v.w;
  }
#pragma unroll
  for (int m = 1; m < 32; m <<= 1) s += __shfl_xor(s, m, 32);
  if (lane == 0) invn[row] = rsqrtf(s);
}

// ---------------------------------------------------------------------------
// Kernel 2: fused GEMM (f16 WMMA, f32 accumulate) + per-block top-4.
// Block stages NROW matching rows as f16 in LDS; each of the 8 waves sweeps
// 256 of the 2048 queries, 2 query-tiles at a time x all 8 n-tiles.
// ---------------------------------------------------------------------------
__global__ void __launch_bounds__(256)
gemm_topk_kernel(const float* __restrict__ Q, const float* __restrict__ M,
                 const float* __restrict__ invn_g,
                 float* __restrict__ pval, int* __restrict__ pidx) {
  extern __shared__ char smem[];
  _Float16* tile    = (_Float16*)smem;
  float*    invnLds = (float*)(smem + (size_t)NROW * LDS_STRIDE * sizeof(_Float16));

  const int n0 = blockIdx.x * NROW;

  // ---- stage matching chunk fp32 -> f16 into LDS (coalesced float4 reads) ----
  for (int i = 0; i < 128; ++i) {
    int flat = threadIdx.x + (i << 8);   // float4 index in chunk (256 per row)
    int r  = flat >> 8;
    int c4 = flat & 255;
    float4 v = ((const float4*)(M + (size_t)(n0 + r) * DIM))[c4];
    v4h h;
    h[0] = (_Float16)v.x; h[1] = (_Float16)v.y;
    h[2] = (_Float16)v.z; h[3] = (_Float16)v.w;
    *(v4h*)(tile + r * LDS_STRIDE + (c4 << 2)) = h;
  }
  if (threadIdx.x < NROW) invnLds[threadIdx.x] = invn_g[n0 + threadIdx.x];
  __syncthreads();

  const int wave = threadIdx.x >> 5;
  const int lane = threadIdx.x & 31;
  const int half = lane >> 4;
  const int lm   = lane & 15;

  float invn[8];
#pragma unroll
  for (int j = 0; j < 8; ++j) invn[j] = invnLds[j * 16 + lm];

  for (int pp = wave; pp < 64; pp += 8) {   // 2 query-tiles (32 rows) per pp
    const int q0 = pp * 32;

    v8f acc[2][8];
#pragma unroll
    for (int t = 0; t < 2; ++t)
#pragma unroll
      for (int j = 0; j < 8; ++j)
#pragma unroll
        for (int e = 0; e < 8; ++e) acc[t][j][e] = 0.f;

#pragma unroll 2
    for (int k0 = 0; k0 < DIM; k0 += 32) {
      // A fragments: 16x32 f16. lane<16: K 0-15; lane>=16: K 16-31
      const float* r0 = Q + (size_t)(q0 + lm) * DIM + k0 + half * 16;
      const float* r1 = r0 + (size_t)16 * DIM;
      float4 x0 = *(const float4*)(r0);
      float4 x1 = *(const float4*)(r0 + 4);
      float4 x2 = *(const float4*)(r0 + 8);
      float4 x3 = *(const float4*)(r0 + 12);
      float4 y0 = *(const float4*)(r1);
      float4 y1 = *(const float4*)(r1 + 4);
      float4 y2 = *(const float4*)(r1 + 8);
      float4 y3 = *(const float4*)(r1 + 12);
      v16h a0, a1;
      a0[0]=(_Float16)x0.x; a0[1]=(_Float16)x0.y; a0[2]=(_Float16)x0.z; a0[3]=(_Float16)x0.w;
      a0[4]=(_Float16)x1.x; a0[5]=(_Float16)x1.y; a0[6]=(_Float16)x1.z; a0[7]=(_Float16)x1.w;
      a0[8]=(_Float16)x2.x; a0[9]=(_Float16)x2.y; a0[10]=(_Float16)x2.z; a0[11]=(_Float16)x2.w;
      a0[12]=(_Float16)x3.x; a0[13]=(_Float16)x3.y; a0[14]=(_Float16)x3.z; a0[15]=(_Float16)x3.w;
      a1[0]=(_Float16)y0.x; a1[1]=(_Float16)y0.y; a1[2]=(_Float16)y0.z; a1[3]=(_Float16)y0.w;
      a1[4]=(_Float16)y1.x; a1[5]=(_Float16)y1.y; a1[6]=(_Float16)y1.z; a1[7]=(_Float16)y1.w;
      a1[8]=(_Float16)y2.x; a1[9]=(_Float16)y2.y; a1[10]=(_Float16)y2.z; a1[11]=(_Float16)y2.w;
      a1[12]=(_Float16)y3.x; a1[13]=(_Float16)y3.y; a1[14]=(_Float16)y3.z; a1[15]=(_Float16)y3.w;

      // B fragments: 32x16 f16. lane = N column; lane<16: K 0-15, lane>=16: K 16-31
      const _Float16* bbase = tile + lm * LDS_STRIDE + k0 + (half << 4);
#pragma unroll
      for (int j = 0; j < 8; ++j) {
        v16h b = *(const v16h*)(bbase + j * 16 * LDS_STRIDE);
        acc[0][j] = __builtin_amdgcn_wmma_f32_16x16x32_f16(
            false, a0, false, b, (short)0, acc[0][j], false, false);
        acc[1][j] = __builtin_amdgcn_wmma_f32_16x16x32_f16(
            false, a1, false, b, (short)0, acc[1][j], false, false);
      }
    }

    // ---- per-query top-4 over this block's 128 columns ----
    // C layout: element v, lane ln -> M = v + 8*(ln>=16), N = ln&15.
#pragma unroll
    for (int t = 0; t < 2; ++t) {
#pragma unroll
      for (int v = 0; v < 8; ++v) {
        float w[8];
#pragma unroll
        for (int j = 0; j < 8; ++j) w[j] = acc[t][j][v] * invn[j];
        const int q = q0 + t * 16 + v + (half << 3);
#pragma unroll
        for (int it = 0; it < TOPK; ++it) {
          float bv = w[0]; int bj = 0;
#pragma unroll
          for (int j = 1; j < 8; ++j)
            if (w[j] > bv) { bv = w[j]; bj = j; }
          int bidx = n0 + bj * 16 + lm;
          // reduce across the 16 lanes of this half (each half = distinct query)
#pragma unroll
          for (int m = 1; m < 16; m <<= 1) {
            float ov = __shfl_xor(bv, m, 16);
            int   oi = __shfl_xor(bidx, m, 16);
            if (ov > bv || (ov == bv && oi < bidx)) { bv = ov; bidx = oi; }
          }
          if (lm == 0) {
            size_t o = ((size_t)q * NBLK + blockIdx.x) * TOPK + it;
            pval[o] = bv;
            pidx[o] = bidx;
          }
          // remove winner from its owning lane
          if ((bidx & 15) == lm) {
            int jj = (bidx - n0) >> 4;
#pragma unroll
            for (int j = 0; j < 8; ++j)
              if (j == jj) w[j] = -__builtin_inff();
          }
        }
      }
    }
  }
}

// ---------------------------------------------------------------------------
// Kernel 3: merge 512 partial top-4 lists per query -> global top-4, then
// gather + average the 4 synth rows. One block per query.
// ---------------------------------------------------------------------------
__device__ __forceinline__ bool better(float v, int i, float V, int I) {
  return (v > V) || (v == V && i < I);
}

__global__ void __launch_bounds__(256)
merge_gather_kernel(const float* __restrict__ pval, const int* __restrict__ pidx,
                    const float* __restrict__ S, float* __restrict__ out) {
  __shared__ float wv[32];
  __shared__ int   wi[32];
  __shared__ int   sel[TOPK];

  const int q    = blockIdx.x;
  const int t    = threadIdx.x;
  const int wave = t >> 5;
  const int lane = t & 31;

  // each thread: sorted top-4 over its 8 candidates
  float lv[4] = {-__builtin_inff(), -__builtin_inff(), -__builtin_inff(), -__builtin_inff()};
  int   li[4] = {0x7fffffff, 0x7fffffff, 0x7fffffff, 0x7fffffff};
  size_t base = (size_t)q * CAND + (size_t)t * 8;
#pragma unroll
  for (int k = 0; k < 8; ++k) {
    float v = pval[base + k];
    int   i = pidx[base + k];
    if (better(v, i, lv[3], li[3])) {
      lv[3] = v; li[3] = i;
#pragma unroll
      for (int m = 3; m >= 1; --m)
        if (better(lv[m], li[m], lv[m - 1], li[m - 1])) {
          float tv = lv[m]; lv[m] = lv[m - 1]; lv[m - 1] = tv;
          int   ti = li[m]; li[m] = li[m - 1]; li[m - 1] = ti;
        }
    }
  }
  // per-wave: extract global top-4 of the wave's 256 candidates
#pragma unroll
  for (int it = 0; it < TOPK; ++it) {
    float bv = lv[0]; int bi = li[0];
#pragma unroll
    for (int m = 1; m < 32; m <<= 1) {
      float ov = __shfl_xor(bv, m, 32);
      int   oi = __shfl_xor(bi, m, 32);
      if (better(ov, oi, bv, bi)) { bv = ov; bi = oi; }
    }
    if (lv[0] == bv && li[0] == bi) {   // owner pops its head
      lv[0] = lv[1]; li[0] = li[1];
      lv[1] = lv[2]; li[1] = li[2];
      lv[2] = lv[3]; li[2] = li[3];
      lv[3] = -__builtin_inff(); li[3] = 0x7fffffff;
    }
    if (lane == 0) { wv[wave * TOPK + it] = bv; wi[wave * TOPK + it] = bi; }
  }
  __syncthreads();
  // wave 0: final top-4 over the 32 wave winners
  if (wave == 0) {
    float fv = wv[lane];
    int   fi = wi[lane];
#pragma unroll
    for (int it = 0; it < TOPK; ++it) {
      float bv = fv; int bi = fi;
#pragma unroll
      for (int m = 1; m < 32; m <<= 1) {
        float ov = __shfl_xor(bv, m, 32);
        int   oi = __shfl_xor(bi, m, 32);
        if (better(ov, oi, bv, bi)) { bv = ov; bi = oi; }
      }
      if (fv == bv && fi == bi) fv = -__builtin_inff();
      if (lane == 0) sel[it] = bi;
    }
  }
  __syncthreads();
  // gather + mean of the 4 selected synth rows
  const float* s0 = S + (size_t)sel[0] * DIM;
  const float* s1 = S + (size_t)sel[1] * DIM;
  const float* s2 = S + (size_t)sel[2] * DIM;
  const float* s3 = S + (size_t)sel[3] * DIM;
  for (int d = t; d < DIM; d += 256)
    out[(size_t)q * DIM + d] = 0.25f * (s0[d] + s1[d] + s2[d] + s3[d]);
}

// ---------------------------------------------------------------------------
extern "C" void kernel_launch(void* const* d_in, const int* in_sizes, int n_in,
                              void* d_out, int out_size, void* d_ws, size_t ws_size,
                              hipStream_t stream) {
  (void)in_sizes; (void)n_in; (void)out_size; (void)ws_size;
  const float* Q = (const float*)d_in[0];
  const float* M = (const float*)d_in[1];
  const float* S = (const float*)d_in[2];
  float* out = (float*)d_out;

  // workspace layout: inv_norms | partial vals | partial idxs  (~34 MB)
  float* invn = (float*)d_ws;
  float* pval = invn + N2;
  int*   pidx = (int*)(pval + (size_t)N1 * CAND);

  size_t dynLds = (size_t)NROW * LDS_STRIDE * sizeof(_Float16) + NROW * sizeof(float);
  hipFuncSetAttribute((const void*)gemm_topk_kernel,
                      hipFuncAttributeMaxDynamicSharedMemorySize, (int)dynLds);

  norms_kernel<<<N2 / 8, 256, 0, stream>>>(M, invn);
  gemm_topk_kernel<<<NBLK, 256, dynLds, stream>>>(Q, M, invn, pval, pidx);
  merge_gather_kernel<<<N1, 256, 0, stream>>>(pval, pidx, S, out);
}
